// FastAttention_71201967833785
// MI455X (gfx1250) — compile-verified, matrix operand order checked by reading, not run on hardware
//
#include <hip/hip_runtime.h>
#include <hip/hip_bf16.h>
#include <stddef.h>

// ---------------------------------------------------------------------------
// FastAttention (SORF linear attention) for gfx1250 / MI455X.
// Contractions on v_wmma_f32_16x16x32_bf16; large GEMMs use a double-buffered
// LDS pipeline fed by GLOBAL_LOAD_ASYNC_TO_LDS_B128 (ASYNCcnt-tracked).
// ---------------------------------------------------------------------------

typedef __attribute__((ext_vector_type(16))) __bf16 v16bf;
typedef __attribute__((ext_vector_type(8)))  float  v8f;
typedef int vi4 __attribute__((vector_size(16)));   // matches builtin param

#define C_DIM 4
#define L_DIM 4096
#define E_DIM 1024
#define H_DIM 16
#define D_DIM 64
#define R_DIM 4
#define P_DIM 64
#define M_FEAT 256
#define NAUG 80

// ---- Tiled GEMM geometry ---------------------------------------------------
#define BM 64
#define BN 128
#define BK 32
#define LDW 36                 // padded LDS row stride (dwords); 144B = 9*16B
#define ASYNC_PER_TILE 6       // async b128 ops issued per thread per tile

// ---- CDNA5 async global->LDS copy (compile-safe guards) --------------------
#if __has_builtin(__builtin_amdgcn_global_load_async_to_lds_b128)
#define HAVE_ASYNC_LDS 1
__device__ __forceinline__ void async_b128(const float* g, float* l) {
  __builtin_amdgcn_global_load_async_to_lds_b128(
      (__attribute__((address_space(1))) vi4*)g,
      (__attribute__((address_space(3))) vi4*)l, 0, 0);
}
#else
#define HAVE_ASYNC_LDS 0
#endif

template <int N>
__device__ __forceinline__ void wait_asynccnt() {
#if HAVE_ASYNC_LDS
#if __has_builtin(__builtin_amdgcn_s_wait_asynccnt)
  __builtin_amdgcn_s_wait_asynccnt(N);
#else
  asm volatile("s_wait_asynccnt %0" ::"i"(N) : "memory");
#endif
#endif
}

// ---- WMMA fragment loaders -------------------------------------------------
// CDNA5 16-bit A 16x32 layout (ISA 7.12.2): lane m (0..15) holds row m,
// K = {hi*8..hi*8+7} and {16+hi*8..16+hi*8+7}; B 32x16 mirrors with lane=col.

__device__ __forceinline__ v16bf frag_row_f32(const float* __restrict__ base,
                                              int ld, int row, int k0, int hi) {
  const float* p0 = base + (size_t)row * ld + k0 + hi * 8;
  const float* p1 = p0 + 16;
  v16bf f;
#pragma unroll
  for (int i = 0; i < 8; ++i) f[i] = (__bf16)p0[i];
#pragma unroll
  for (int i = 0; i < 8; ++i) f[8 + i] = (__bf16)p1[i];
  return f;
}

__device__ __forceinline__ v16bf frag_row_bf16(const __bf16* __restrict__ base,
                                               int ld, int row, int k0, int hi) {
  const __bf16* p0 = base + (size_t)row * ld + k0 + hi * 8;
  const __bf16* p1 = p0 + 16;
  v16bf f;
#pragma unroll
  for (int i = 0; i < 8; ++i) f[i] = p0[i];
#pragma unroll
  for (int i = 0; i < 8; ++i) f[8 + i] = p1[i];
  return f;
}

__device__ __forceinline__ v16bf frag_col_f32(const float* __restrict__ colbase,
                                              int ld, int k0, int hi) {
  v16bf f;
#pragma unroll
  for (int i = 0; i < 8; ++i)
    f[i] = (__bf16)colbase[(size_t)(k0 + hi * 8 + i) * ld];
#pragma unroll
  for (int i = 0; i < 8; ++i)
    f[8 + i] = (__bf16)colbase[(size_t)(k0 + 16 + hi * 8 + i) * ld];
  return f;
}

// LDS tile (f32, padded stride LDW), convert to bf16 at read time.
__device__ __forceinline__ v16bf frag_lds(const float* tile, int row, int kk, int hi) {
  const float* p0 = tile + row * LDW + kk + hi * 8;
  const float* p1 = p0 + 16;
  v16bf f;
#pragma unroll
  for (int i = 0; i < 8; ++i) f[i] = (__bf16)p0[i];
#pragma unroll
  for (int i = 0; i < 8; ++i) f[8 + i] = (__bf16)p1[i];
  return f;
}

// ---- K1 / K5: Out(MxN) = A(MxK) @ W(NxK)^T ---------------------------------
// Block: 256 threads (8 waves) -> 64x128 tile. Each wave: 32x32 (2x2 wmma).
// Double-buffered LDS, async global->LDS b128 copies, s_wait_asynccnt pipeline.

__device__ __forceinline__ void stage_tile(const float* __restrict__ A,
                                           const float* __restrict__ W,
                                           float* Abuf, float* Bbuf,
                                           int m_blk, int n_blk, int K, int k0,
                                           int tid) {
  int ar = tid >> 2, ac = (tid & 3) * 8;    // A: 64 rows x 32 cols, 8 f32/thr
  const float* ga = A + (size_t)(m_blk + ar) * K + k0 + ac;
  float* la = Abuf + ar * LDW + ac;
  int br = tid >> 1, bc = (tid & 1) * 16;   // B: 128 rows x 32 cols, 16 f32/thr
  const float* gb = W + (size_t)(n_blk + br) * K + k0 + bc;
  float* lb = Bbuf + br * LDW + bc;
#if HAVE_ASYNC_LDS
#pragma unroll
  for (int j = 0; j < 2; ++j) async_b128(ga + j * 4, la + j * 4);
#pragma unroll
  for (int j = 0; j < 4; ++j) async_b128(gb + j * 4, lb + j * 4);
#else
#pragma unroll
  for (int j = 0; j < 2; ++j)
    *(float4*)(la + j * 4) = *(const float4*)(ga + j * 4);
#pragma unroll
  for (int j = 0; j < 4; ++j)
    *(float4*)(lb + j * 4) = *(const float4*)(gb + j * 4);
#endif
}

__global__ void __launch_bounds__(256)
gemm_xwT(const float* __restrict__ A, const float* __restrict__ W,
         float* __restrict__ Out, int M, int N, int K) {
  __shared__ float Asmem[2][BM * LDW];
  __shared__ float Bsmem[2][BN * LDW];

  int tid = threadIdx.x;
  int wave = tid >> 5;
  int lane = tid & 31;
  int hi = lane >> 4;
  int nl = lane & 15;

  int nb = N / BN;
  int m_blk = (blockIdx.x / nb) * BM;
  int n_blk = (blockIdx.x % nb) * BN;

  // wave -> 2x2 grid of 16x16 tiles: rows (wm..wm+1), cols (wn..wn+1)
  int wm = (wave >> 2) * 2;   // 0 or 2  (m-tiles 0..3)
  int wn = (wave & 3) * 2;    // 0,2,4,6 (n-tiles 0..7)

  v8f acc[2][2] = {};

  int nk = K / BK;
  stage_tile(A, W, Asmem[0], Bsmem[0], m_blk, n_blk, K, 0, tid);

  for (int t = 0; t < nk; ++t) {
    int cur = t & 1;
    if (t + 1 < nk) {
      stage_tile(A, W, Asmem[cur ^ 1], Bsmem[cur ^ 1], m_blk, n_blk, K,
                 (t + 1) * BK, tid);
      wait_asynccnt<ASYNC_PER_TILE>();   // our copies for tile t are done
    } else {
      wait_asynccnt<0>();
    }
    __syncthreads();

    const float* At = Asmem[cur];
    const float* Bt = Bsmem[cur];
    v16bf a0 = frag_lds(At, (wm + 0) * 16 + nl, 0, hi);
    v16bf a1 = frag_lds(At, (wm + 1) * 16 + nl, 0, hi);
    v16bf b0 = frag_lds(Bt, (wn + 0) * 16 + nl, 0, hi);
    v16bf b1 = frag_lds(Bt, (wn + 1) * 16 + nl, 0, hi);
    acc[0][0] = __builtin_amdgcn_wmma_f32_16x16x32_bf16(false, a0, false, b0,
                                                        (short)0, acc[0][0],
                                                        false, false);
    acc[0][1] = __builtin_amdgcn_wmma_f32_16x16x32_bf16(false, a0, false, b1,
                                                        (short)0, acc[0][1],
                                                        false, false);
    acc[1][0] = __builtin_amdgcn_wmma_f32_16x16x32_bf16(false, a1, false, b0,
                                                        (short)0, acc[1][0],
                                                        false, false);
    acc[1][1] = __builtin_amdgcn_wmma_f32_16x16x32_bf16(false, a1, false, b1,
                                                        (short)0, acc[1][1],
                                                        false, false);
    __syncthreads();
  }

#pragma unroll
  for (int mi = 0; mi < 2; ++mi)
#pragma unroll
    for (int ni = 0; ni < 2; ++ni)
#pragma unroll
      for (int r = 0; r < 8; ++r)
        Out[(size_t)(m_blk + (wm + mi) * 16 + r + hi * 8) * N + n_blk +
            (wn + ni) * 16 + nl] = acc[mi][ni][r];
}

// ---- K2: RoPE + SORF phi (register-resident FWHT-64) -----------------------
__device__ __forceinline__ void fwht64(float* t) {
#pragma unroll
  for (int s = 1; s < 64; s <<= 1) {
#pragma unroll
    for (int i = 0; i < 64; ++i) {
      if ((i & s) == 0) {
        float a = t[i], b = t[i + s];
        t[i] = a + b;
        t[i + s] = a - b;
      }
    }
  }
}

__global__ void rope_sorf_phi(const float* __restrict__ qkv,
                              const float* __restrict__ signs,
                              const float* __restrict__ scales,
                              __bf16* __restrict__ q_phi,     // (C*H, L, 256)
                              __bf16* __restrict__ k_phiT) {  // (C*H, 256, L)
  int idx = blockIdx.x * blockDim.x + threadIdx.x;
  int l = idx % L_DIM;
  int h = (idx / L_DIM) % H_DIM;
  int c = (idx / (L_DIM * H_DIM)) % C_DIM;
  int which = idx / (L_DIM * H_DIM * C_DIM);  // 0 = q, 1 = k

  const float* src =
      qkv + (size_t)(c * L_DIM + l) * (3 * E_DIM) + which * E_DIM + h * D_DIM;
  float row[64];
#pragma unroll
  for (int i = 0; i < 64; ++i) row[i] = src[i];

  if (l >= 1) {
    float pos = (float)(l - 1);
#pragma unroll
    for (int i = 0; i < 32; ++i) {
      float fr = __expf(-(2.0f * i / 64.0f) * 9.210340371976184f);
      float ang = pos * fr;
      float cs = __cosf(ang), sn = __sinf(ang);
      float t1 = row[2 * i], t2 = row[2 * i + 1];
      row[2 * i]     = t1 * cs - t2 * sn;
      row[2 * i + 1] = t1 * sn + t2 * cs;
    }
  }

  float nsq = 0.0f;
#pragma unroll
  for (int i = 0; i < 64; ++i) nsq += row[i] * row[i];
  nsq *= 0.5f;

  int ch = c * H_DIM + h;
  for (int r = 0; r < R_DIM; ++r) {
    const float* s1 = signs + (((size_t)h * R_DIM + r) * 3 + 0) * P_DIM;
    const float* s2 = s1 + P_DIM;
    const float* s3 = s2 + P_DIM;
    const float* g  = scales + ((size_t)h * R_DIM + r) * P_DIM;
    float t[64];
#pragma unroll
    for (int i = 0; i < 64; ++i) t[i] = row[i] * s1[i];
    fwht64(t);
#pragma unroll
    for (int i = 0; i < 64; ++i) t[i] = t[i] * 0.125f * s2[i];
    fwht64(t);
#pragma unroll
    for (int i = 0; i < 64; ++i) t[i] = t[i] * 0.125f * s3[i];
    fwht64(t);
#pragma unroll
    for (int i = 0; i < 64; ++i) {
      float phi = t[i] * 0.125f * g[i] * 0.125f;
      float val = __expf(phi - nsq) * 0.0625f;
      if (which == 0)
        q_phi[((size_t)ch * L_DIM + l) * M_FEAT + r * P_DIM + i] = (__bf16)val;
      else
        k_phiT[((size_t)ch * M_FEAT + r * P_DIM + i) * L_DIM + l] = (__bf16)val;
    }
  }
}

// ---- K3: S_aug(256x80) = k_phi^T @ [v | 1 | 0..] per (c,h) -----------------
__global__ void kv_state(const __bf16* __restrict__ k_phiT,
                         const float* __restrict__ qkv,
                         float* __restrict__ S_aug) {
  int wave = (int)((blockIdx.x * blockDim.x + threadIdx.x) >> 5);
  int lane = threadIdx.x & 31;
  int hi = lane >> 4, nl = lane & 15;
  int nt = wave % 5;
  int mt = (wave / 5) % 16;
  int ch = wave / 80;
  int c = ch >> 4, h = ch & 15;

  const __bf16* Abase = k_phiT + (size_t)ch * M_FEAT * L_DIM;
  const float* vcol = qkv + (size_t)c * L_DIM * (3 * E_DIM) + 2 * E_DIM +
                      h * D_DIM + (nt < 4 ? nt * 16 + nl : 0);
  v16bf bconst;
  {
    __bf16 v1 = (__bf16)((nt == 4 && nl == 0) ? 1.0f : 0.0f);
#pragma unroll
    for (int i = 0; i < 16; ++i) bconst[i] = v1;
  }

  v8f acc = {};
  for (int k0 = 0; k0 < L_DIM; k0 += 32) {
    v16bf a = frag_row_bf16(Abase, L_DIM, mt * 16 + nl, k0, hi);
    v16bf b = (nt < 4) ? frag_col_f32(vcol, 3 * E_DIM, k0, hi) : bconst;
    acc = __builtin_amdgcn_wmma_f32_16x16x32_bf16(false, a, false, b, (short)0,
                                                  acc, false, false);
  }
  float* outb = S_aug + (size_t)ch * M_FEAT * NAUG;
#pragma unroll
  for (int r = 0; r < 8; ++r)
    outb[(size_t)(mt * 16 + r + hi * 8) * NAUG + nt * 16 + nl] = acc[r];
}

// ---- K4: out_aug(Lx80) = q_phi @ S_aug; divide by denom (col 64) -----------
__global__ void attn_apply(const __bf16* __restrict__ q_phi,
                           const float* __restrict__ S_aug,
                           float* __restrict__ out_heads) {  // (C*L, 1024)
  int wave = (int)((blockIdx.x * blockDim.x + threadIdx.x) >> 5);
  int lane = threadIdx.x & 31;
  int hi = lane >> 4, nl = lane & 15;
  int lt = wave % (L_DIM / 16);
  int ch = wave / (L_DIM / 16);
  int c = ch >> 4, h = ch & 15;

  const __bf16* Abase = q_phi + (size_t)ch * L_DIM * M_FEAT;
  const float* Sb = S_aug + (size_t)ch * M_FEAT * NAUG;

  v8f acc[5] = {};
  for (int k0 = 0; k0 < M_FEAT; k0 += 32) {
    v16bf a = frag_row_bf16(Abase, M_FEAT, lt * 16 + nl, k0, hi);
#pragma unroll
    for (int s = 0; s < 5; ++s) {
      v16bf b = frag_col_f32(Sb + s * 16 + nl, NAUG, k0, hi);
      acc[s] = __builtin_amdgcn_wmma_f32_16x16x32_bf16(false, a, false, b,
                                                       (short)0, acc[s], false,
                                                       false);
    }
  }
#pragma unroll
  for (int r = 0; r < 8; ++r) {
    float den = __shfl(acc[4][r], lane & 16, 32);
    den = fmaxf(den, 1e-6f);
    float inv = 1.0f / den;
#pragma unroll
    for (int s = 0; s < 4; ++s) {
      size_t o = ((size_t)(c * L_DIM) + lt * 16 + r + hi * 8) * E_DIM +
                 h * D_DIM + s * 16 + nl;
      out_heads[o] = acc[s][r] * inv;
    }
  }
}

// ---------------------------------------------------------------------------
extern "C" void kernel_launch(void* const* d_in, const int* in_sizes, int n_in,
                              void* d_out, int out_size, void* d_ws, size_t ws_size,
                              hipStream_t stream) {
  const float* x      = (const float*)d_in[0];
  const float* w_qkv  = (const float*)d_in[1];
  const float* w_out  = (const float*)d_in[2];
  const float* signs  = (const float*)d_in[3];
  const float* scales = (const float*)d_in[4];

  char* ws = (char*)d_ws;
  size_t off = 0;
  float* qkv = (float*)(ws + off);        off += (size_t)16384 * 3072 * 4;     // 192 MB
  __bf16* q_phi = (__bf16*)(ws + off);    off += (size_t)64 * 4096 * 256 * 2;  // 128 MB
  __bf16* k_phiT = (__bf16*)(ws + off);   off += (size_t)64 * 256 * 4096 * 2;  // 128 MB
  float* S_aug = (float*)(ws + off);      off += (size_t)64 * 256 * NAUG * 4;  // 5 MB
  float* out_heads = (float*)(ws + off);  off += (size_t)16384 * 1024 * 4;     // 64 MB
  float* out = (float*)d_out;

  // K1: qkv = x @ w_qkv^T. (16384/64) x (3072/128) = 256 x 24 blocks.
  gemm_xwT<<<6144, 256, 0, stream>>>(x, w_qkv, qkv, 16384, 3072, 1024);
  // K2: RoPE + SORF phi. 2*C*H*L = 524288 threads.
  rope_sorf_phi<<<2048, 256, 0, stream>>>(qkv, signs, scales, q_phi, k_phiT);
  // K3: S_aug per (c,h). 64 * 16 * 5 = 5120 waves.
  kv_state<<<640, 256, 0, stream>>>(k_phiT, qkv, S_aug);
  // K4: q_phi @ S_aug + normalize. 64 * 256 = 16384 waves.
  attn_apply<<<2048, 256, 0, stream>>>(q_phi, S_aug, out_heads);
  // K5: final projection. (16384/64) x (1024/128) = 256 x 8 blocks.
  gemm_xwT<<<2048, 256, 0, stream>>>(out_heads, w_out, out, 16384, 1024, 1024);
}
